// GNN_9895604650393
// MI455X (gfx1250) — matrix operand order repaired and checked by reading, not compiled, not run on hardware
//
#include <hip/hip_runtime.h>
#include <hip/hip_bf16.h>

// ---------- types ----------
typedef __attribute__((ext_vector_type(16))) __bf16 v16bf;
typedef __attribute__((ext_vector_type(8)))  float  v8f;

union Frag16 {
    v16bf v;
    uint4 u[2];
};

// ---------- helpers ----------
__device__ __forceinline__ unsigned short f2bf(float f) {
    unsigned int u = __builtin_bit_cast(unsigned int, f);
    u += 0x7FFFu + ((u >> 16) & 1u);            // round-to-nearest-even
    return (unsigned short)(u >> 16);
}
__device__ __forceinline__ unsigned int pk2(float a, float b) {
    return (unsigned int)f2bf(a) | ((unsigned int)f2bf(b) << 16);
}
__device__ __forceinline__ uint4 pk8(float4 a, float4 b) {
    return make_uint4(pk2(a.x, a.y), pk2(a.z, a.w), pk2(b.x, b.y), pk2(b.z, b.w));
}

// A-fragment (16x32 bf16) built from an fp32 row in global memory.
__device__ __forceinline__ void loadAfragF32(const float* __restrict__ rowp, int k0, Frag16& f) {
    float4 f0 = *(const float4*)(rowp + k0);
    float4 f1 = *(const float4*)(rowp + k0 + 4);
    float4 f2 = *(const float4*)(rowp + k0 + 16);
    float4 f3 = *(const float4*)(rowp + k0 + 20);
    f.u[0] = pk8(f0, f1);
    f.u[1] = pk8(f2, f3);
}

// A-fragment from a bf16 row (global or LDS): two raw 16-B loads, no conversion.
__device__ __forceinline__ void loadAfragBf(const unsigned short* rowp, int k0, Frag16& f) {
    f.u[0] = *(const uint4*)(rowp + k0);
    f.u[1] = *(const uint4*)(rowp + k0 + 16);
}

// B-fragment (32x16 bf16) from W^T staged in LDS: two 16-B loads at idx and idx+16.
__device__ __forceinline__ void loadBfrag(const unsigned short* Wt, int idx, Frag16& f) {
    const uint4* p = (const uint4*)(Wt + idx);
    f.u[0] = p[0];
    f.u[1] = p[2];
}

__device__ __forceinline__ v8f wmma_bf16(const Frag16& a, const Frag16& b, v8f c) {
    return __builtin_amdgcn_wmma_f32_16x16x32_bf16(false, a.v, false, b.v,
                                                   (short)0, c, false, false);
}

// ---------- degree / norm ----------
__global__ void __launch_bounds__(256) deg_init(float* deg, int N) {
    int i = blockIdx.x * 256 + threadIdx.x;
    if (i < N) deg[i] = 1.0f;                  // self loop
}
__global__ void __launch_bounds__(256) deg_count(float* deg, const int* __restrict__ dst, int E) {
    int i = blockIdx.x * 256 + threadIdx.x;
    if (i < E) atomicAdd(&deg[dst[i]], 1.0f);
}
__global__ void __launch_bounds__(256) deg_fin(float* deg, int N) {
    int i = blockIdx.x * 256 + threadIdx.x;
    if (i < N) deg[i] = rsqrtf(deg[i]);        // deg >= 1 always
}

// ---------- ReLU + fp32->bf16 pack (one pass per layer, 2 values/thread) ----------
__global__ void __launch_bounds__(256)
cvt_relu(const float2* __restrict__ in, unsigned int* __restrict__ outp, int n2) {
    int i = blockIdx.x * 256 + threadIdx.x;
    if (i < n2) {
        float2 v = in[i];
        outp[i] = pk2(fmaxf(v.x, 0.f), fmaxf(v.y, 0.f));
    }
}

// ---------- node GEMM: Y[N,64] = X[N,K] @ W[K,64] ----------
// BF16IN=0: X fp32 (converted on load); BF16IN=1: X bf16 rows.
// Persistent waves: B-fragments hoisted to registers, tile-stride loop.
template <int K, int BF16IN>
__global__ void __launch_bounds__(256)
gcn_gemm(const void* __restrict__ Xv, const float* __restrict__ W,
         float* __restrict__ Y, int N) {
    __shared__ unsigned short Wt[64 * K];      // W^T, bf16
    int tid = threadIdx.x;
    for (int i = tid; i < K * 64; i += 256) {
        int k = i >> 6, n = i & 63;
        Wt[n * K + k] = f2bf(W[i]);
    }
    __syncthreads();

    int wave = tid >> 5, lane = tid & 31;
    int m = lane & 15, h = lane >> 4, mb = h * 8;
    constexpr int NKT = K / 32;

    // hoist all B-fragments (loop-invariant weights) into registers
    Frag16 bf_[4][NKT];
#pragma unroll
    for (int nt = 0; nt < 4; ++nt)
#pragma unroll
        for (int kt = 0; kt < NKT; ++kt)
            loadBfrag(Wt, (nt * 16 + m) * K + kt * 32 + h * 8, bf_[nt][kt]);

    int nTiles = (N + 15) >> 4;
    int stride = gridDim.x * 8;
    for (int tile = blockIdx.x * 8 + wave; tile < nTiles; tile += stride) {
        int row = tile * 16 + m;
        if (row >= N) row = N - 1;

        Frag16 af[NKT];
        if (BF16IN) {
            const unsigned short* xrow = (const unsigned short*)Xv + (size_t)row * K;
#pragma unroll
            for (int kt = 0; kt < NKT; ++kt)
                loadAfragBf(xrow, kt * 32 + h * 8, af[kt]);
        } else {
            const float* xrow = (const float*)Xv + (size_t)row * K;
#pragma unroll
            for (int kt = 0; kt < NKT; ++kt)
                loadAfragF32(xrow, kt * 32 + h * 8, af[kt]);
        }

#pragma unroll
        for (int nt = 0; nt < 4; ++nt) {
            v8f acc = {0.f, 0.f, 0.f, 0.f, 0.f, 0.f, 0.f, 0.f};
#pragma unroll
            for (int kt = 0; kt < NKT; ++kt)
                acc = wmma_bf16(af[kt], bf_[nt][kt], acc);
            int n = nt * 16 + m;
#pragma unroll
            for (int v = 0; v < 8; ++v) {
                int r = tile * 16 + mb + v;
                if (r < N) Y[(size_t)r * 64 + n] = acc[v];
            }
        }
    }
}

// ---------- aggregation ----------
// agg[i,f] = b[f] + pre[i,f]*dinv[i]^2   (bias + self-loop message)
__global__ void __launch_bounds__(256)
agg_init(const float* __restrict__ pre, const float* __restrict__ dinv,
         const float* __restrict__ b, float* __restrict__ agg, int N) {
    int i = blockIdx.x * 256 + threadIdx.x;
    if (i < N * 64) {
        int node = i >> 6, f = i & 63;
        float dv = dinv[node];
        agg[i] = b[f] + pre[i] * dv * dv;
    }
}
// agg[dst] += pre[src] * dinv[src]*dinv[dst]   (4 floats per thread)
__global__ void __launch_bounds__(256)
agg_scatter(const float* __restrict__ pre, const float* __restrict__ dinv,
            const int* __restrict__ src, const int* __restrict__ dst,
            float* __restrict__ agg, int E) {
    int idx = blockIdx.x * 256 + threadIdx.x;
    if (idx < E * 16) {
        int e = idx >> 4, f = (idx & 15) << 2;
        int s = src[e], d = dst[e];
        float nrm = dinv[s] * dinv[d];
        float4 hv = *(const float4*)(pre + (size_t)s * 64 + f);
        float* ap = agg + (size_t)d * 64 + f;
        atomicAdd(ap + 0, hv.x * nrm);
        atomicAdd(ap + 1, hv.y * nrm);
        atomicAdd(ap + 2, hv.z * nrm);
        atomicAdd(ap + 3, hv.w * nrm);
    }
}

// ---------- edge MLP: out[e,:] = relu([h2[src],h2[dst]] @ Wm1 + bm1) @ Wm2 + bm2 ----------
// H is bf16 (already ReLU'd). Persistent waves; all weight fragments in registers;
// biases folded into the WMMA accumulator initialization (D = A*B + C).
__global__ void __launch_bounds__(256)
edge_mlp(const unsigned short* __restrict__ H,
         const int* __restrict__ src, const int* __restrict__ dst,
         const float* __restrict__ Wm1, const float* __restrict__ bm1,
         const float* __restrict__ Wm2, const float* __restrict__ bm2,
         float* __restrict__ out, int E) {
    __shared__ unsigned short W1t[64 * 128];   // Wm1^T bf16 (16 KB)
    __shared__ unsigned short W2t[16 * 64];    // Wm2^T bf16 (2 KB)
    __shared__ unsigned short Zt[8][16 * 64];  // per-wave z tile, bf16 (16 KB)

    int tid = threadIdx.x;
    for (int i = tid; i < 128 * 64; i += 256) {
        int k = i >> 6, n = i & 63;
        W1t[n * 128 + k] = f2bf(Wm1[i]);
    }
    for (int i = tid; i < 64 * 16; i += 256) {
        int k = i >> 4, n = i & 15;
        W2t[n * 64 + k] = f2bf(Wm2[i]);
    }
    __syncthreads();

    int wave = tid >> 5, lane = tid & 31;
    int m = lane & 15, h = lane >> 4, mb = h * 8;

    // hoist all weight fragments into registers (loop-invariant)
    Frag16 b1f[4][4];
#pragma unroll
    for (int nt = 0; nt < 4; ++nt)
#pragma unroll
        for (int kt = 0; kt < 4; ++kt)
            loadBfrag(W1t, (nt * 16 + m) * 128 + kt * 32 + h * 8, b1f[nt][kt]);
    Frag16 b2f[2];
#pragma unroll
    for (int kt = 0; kt < 2; ++kt)
        loadBfrag(W2t, m * 64 + kt * 32 + h * 8, b2f[kt]);

    float bias1[4];
#pragma unroll
    for (int nt = 0; nt < 4; ++nt) bias1[nt] = bm1[nt * 16 + m];
    float bias2 = bm2[m];

    unsigned short* zrow = Zt[wave];
    int nTiles = (E + 15) >> 4;
    int stride = gridDim.x * 8;
    for (int tile = blockIdx.x * 8 + wave; tile < nTiles; tile += stride) {
        int e = tile * 16 + m;
        if (e >= E) e = E - 1;
        const unsigned short* srow = H + (size_t)src[e] * 64;
        const unsigned short* drow = H + (size_t)dst[e] * 64;

        // A = ef tile (16 x 128): K 0..63 from h2[src], 64..127 from h2[dst].
        Frag16 af[4];
#pragma unroll
        for (int kt = 0; kt < 4; ++kt)
            loadAfragBf((kt < 2) ? srow : drow, (kt & 1) * 32 + h * 8, af[kt]);

        // z = ef @ Wm1 + bm1 : bias pre-loaded into C, 4x4 WMMA from registers
        v8f acc[4];
#pragma unroll
        for (int nt = 0; nt < 4; ++nt) {
            float b_ = bias1[nt];
            acc[nt] = {b_, b_, b_, b_, b_, b_, b_, b_};
#pragma unroll
            for (int kt = 0; kt < 4; ++kt)
                acc[nt] = wmma_bf16(af[kt], b1f[nt][kt], acc[nt]);
        }

        // ReLU, transpose through LDS (C/D layout -> A layout)
#pragma unroll
        for (int nt = 0; nt < 4; ++nt) {
            int n = nt * 16 + m;
#pragma unroll
            for (int v = 0; v < 8; ++v) {
                float z = fmaxf(acc[nt][v], 0.f);
                zrow[(mb + v) * 64 + n] = f2bf(z);
            }
        }

        // out = z @ Wm2 + bm2 : bias pre-loaded into C, 2 K-tiles of WMMA
        v8f acc2 = {bias2, bias2, bias2, bias2, bias2, bias2, bias2, bias2};
#pragma unroll
        for (int kt = 0; kt < 2; ++kt) {
            Frag16 a2;
            loadAfragBf(zrow + m * 64, kt * 32 + h * 8, a2);
            acc2 = wmma_bf16(a2, b2f[kt], acc2);
        }

#pragma unroll
        for (int v = 0; v < 8; ++v) {
            int r = tile * 16 + mb + v;
            if (r < E) out[(size_t)r * 16 + m] = acc2[v];
        }
    }
}

// ---------- host launch ----------
extern "C" void kernel_launch(void* const* d_in, const int* in_sizes, int n_in,
                              void* d_out, int out_size, void* d_ws, size_t ws_size,
                              hipStream_t stream) {
    (void)n_in; (void)out_size; (void)ws_size;
    const float* x   = (const float*)d_in[0];
    const int*   ei  = (const int*)d_in[1];
    const float* W1  = (const float*)d_in[2];
    const float* b1  = (const float*)d_in[3];
    const float* W2  = (const float*)d_in[4];
    const float* b2  = (const float*)d_in[5];
    const float* Wm1 = (const float*)d_in[6];
    const float* bm1 = (const float*)d_in[7];
    const float* Wm2 = (const float*)d_in[8];
    const float* bm2 = (const float*)d_in[9];

    const int N = in_sizes[0] / 128;
    const int E = in_sizes[1] / 2;
    const int* src = ei;
    const int* dst = ei + E;

    float* ws   = (float*)d_ws;
    float* dinv = ws;
    size_t off  = ((size_t)N + 63) & ~(size_t)63;
    float* bufA = ws + off;                           // pre-aggregation features (fp32)
    float* bufB = ws + off + (size_t)N * 64;          // aggregated features (fp32)
    unsigned short* hbf =
        (unsigned short*)(ws + off + (size_t)2 * N * 64);  // relu'd bf16 features

    const int nGridN  = (N + 255) / 256;
    const int nGridE  = (E + 255) / 256;
    const int nGridNF = (N * 64 + 255) / 256;
    const int nGridC  = (N * 32 + 255) / 256;         // cvt_relu: 2 values/thread
    const int nGridEF = (E * 16 + 255) / 256;
    const int nTilesN = (N + 15) / 16;
    const int nTilesE = (E + 15) / 16;
    int gGemm = (nTilesN + 7) / 8; if (gGemm > 320) gGemm = 320;   // persistent
    int gEdge = (nTilesE + 7) / 8; if (gEdge > 640) gEdge = 640;   // persistent

    // degrees -> dinv
    deg_init <<<nGridN, 256, 0, stream>>>(dinv, N);
    deg_count<<<nGridE, 256, 0, stream>>>(dinv, dst, E);
    deg_fin  <<<nGridN, 256, 0, stream>>>(dinv, N);

    // layer 1: pre = x @ W1 ; agg = b1 + self-loop + scatter ; h1 = relu->bf16
    gcn_gemm<128, 0><<<gGemm, 256, 0, stream>>>(x, W1, bufA, N);
    agg_init   <<<nGridNF, 256, 0, stream>>>(bufA, dinv, b1, bufB, N);
    agg_scatter<<<nGridEF, 256, 0, stream>>>(bufA, dinv, src, dst, bufB, E);
    cvt_relu   <<<nGridC, 256, 0, stream>>>((const float2*)bufB, (unsigned int*)hbf, N * 32);

    // layer 2: pre = h1 @ W2 ; agg = b2 + self-loop + scatter ; h2 = relu->bf16
    gcn_gemm<64, 1><<<gGemm, 256, 0, stream>>>(hbf, W2, bufA, N);
    agg_init   <<<nGridNF, 256, 0, stream>>>(bufA, dinv, b2, bufB, N);
    agg_scatter<<<nGridEF, 256, 0, stream>>>(bufA, dinv, src, dst, bufB, E);
    cvt_relu   <<<nGridC, 256, 0, stream>>>((const float2*)bufB, (unsigned int*)hbf, N * 32);

    // edge MLP on bf16 features
    edge_mlp<<<gEdge, 256, 0, stream>>>(hbf, src, dst, Wm1, bm1, Wm2, bm2,
                                        (float*)d_out, E);
}
